// GRU_pytorch_53455162966388
// MI455X (gfx1250) — matrix operand order repaired and checked by reading, not compiled, not run on hardware
//
#include <hip/hip_runtime.h>
#include <hip/hip_bf16.h>

// ---------------------------------------------------------------------------
// GRU on gfx1250 (MI455X): bf16 WMMA for both the batched input projection
// and the per-timestep recurrent GEMMs. Wave32, 16x16x32 bf16 -> f32 tiles.
// x / inp_W / hid_W are pre-converted to bf16 (weights transposed to [N x K])
// so the GEMM inner loops are pure global_load_b128 + v_wmma.
// ---------------------------------------------------------------------------

typedef __attribute__((ext_vector_type(16))) __bf16 v16bf;
typedef __attribute__((ext_vector_type(8)))  float  v8f;

#define B_   64
#define S_   512
#define I_   1024
#define H_   1024
#define G_   3072          // 3*H
#define MTOT (B_ * S_)     // 32768 rows of the input projection

// round-to-nearest-even f32 -> bf16 (bit pattern)
__device__ __forceinline__ unsigned short f32_to_bf16_bits(float f) {
  union { float f; unsigned u; } a;
  a.f = f;
  unsigned u = a.u;
  unsigned r = u + 0x7FFFu + ((u >> 16) & 1u);
  return (unsigned short)(r >> 16);
}

union FragBF {
  v16bf v;
  uint4 u[2];
  unsigned short s[16];
};

__device__ __forceinline__ v8f wmma_bf16(v16bf a, v16bf b, v8f c) {
  // (neg_a, A, neg_b, B, c_mod, C, reuse_a, reuse_b)
  return __builtin_amdgcn_wmma_f32_16x16x32_bf16(false, a, false, b,
                                                 (short)0, c, false, false);
}

// A-matrix 16x32 bf16 fragment from bf16 row-major [rows x ld] memory.
// lane<16: row = base row + (lane&15), K = {0..7, 16..23}
// lane>=16: same row set, K = {8..15, 24..31}
__device__ __forceinline__ v16bf load_a_bf(const unsigned short* __restrict__ X,
                                           int row, int ld, int k0, int lane) {
  const int half = lane >> 4;
  const unsigned short* p = X + (size_t)row * ld + k0 + half * 8;
  FragBF fr;
  fr.u[0] = *(const uint4*)(p);        // K +0..7   (or +8..15)
  fr.u[1] = *(const uint4*)(p + 16);   // K +16..23 (or +24..31)
  return fr.v;
}

// B-matrix 32x16 bf16 fragment from transposed weights Wt [N x K] row-major.
// lane<16: col N = base + (lane&15), K = {0..7, 16..23} ; lane>=16: K = {8..15, 24..31}
__device__ __forceinline__ v16bf load_b_bf(const unsigned short* __restrict__ Wt,
                                           int col, int ld, int k0, int lane) {
  const int khalf = lane >> 4;
  const unsigned short* p = Wt + (size_t)col * ld + k0 + khalf * 8;
  FragBF fr;
  fr.u[0] = *(const uint4*)(p);
  fr.u[1] = *(const uint4*)(p + 16);
  return fr.v;
}

// ---------------------------------------------------------------------------
// Kernel: transpose + convert weights W[K x N] (f32) -> Wt[N x K] (bf16)
// ---------------------------------------------------------------------------
__global__ void k_transpose_bf16(const float* __restrict__ W,
                                 unsigned short* __restrict__ Wt,
                                 int K, int N) {
  int idx = blockIdx.x * blockDim.x + threadIdx.x;
  if (idx >= K * N) return;
  int n = idx / K;
  int k = idx - n * K;
  Wt[idx] = f32_to_bf16_bits(W[(size_t)k * N + n]);
}

// Kernel: straight convert x (f32, row-major, K contiguous) -> bf16.
// 8 elements per thread, fully coalesced.
__global__ void k_convert_bf16_vec8(const float* __restrict__ X,
                                    unsigned short* __restrict__ Xb,
                                    int n8) {
  int i = blockIdx.x * blockDim.x + threadIdx.x;
  if (i >= n8) return;
  const float4 f0 = *(const float4*)(X + (size_t)i * 8);
  const float4 f1 = *(const float4*)(X + (size_t)i * 8 + 4);
  FragBF fr;
  fr.s[0] = f32_to_bf16_bits(f0.x); fr.s[1] = f32_to_bf16_bits(f0.y);
  fr.s[2] = f32_to_bf16_bits(f0.z); fr.s[3] = f32_to_bf16_bits(f0.w);
  fr.s[4] = f32_to_bf16_bits(f1.x); fr.s[5] = f32_to_bf16_bits(f1.y);
  fr.s[6] = f32_to_bf16_bits(f1.z); fr.s[7] = f32_to_bf16_bits(f1.w);
  *(uint4*)(Xb + (size_t)i * 8) = fr.u[0];
}

// Zero-init the h0 state (f32 + bf16 views)
__global__ void k_zero_state(float* __restrict__ hf,
                             unsigned short* __restrict__ hb, int n) {
  int i = blockIdx.x * blockDim.x + threadIdx.x;
  if (i < n) { hf[i] = 0.0f; hb[i] = 0; }
}

__global__ void k_copy_f32(const float* __restrict__ src,
                           float* __restrict__ dst, int n) {
  int i = blockIdx.x * blockDim.x + threadIdx.x;
  if (i < n) dst[i] = src[i];
}

// ---------------------------------------------------------------------------
// Phase 1: gates_x[MTOT x G] = Xb[MTOT x I] * inp_Wt^T + inp_b
// Block: 256 thr = 8 waves in 4(M) x 2(N); per-wave 32(M) x 64(N) tile
// (2x4 WMMA accumulators, A fragments reused over 4 B fragments).
// Block tile: 128(M) x 128(N). Grid: (MTOT/128, G/128).
// ---------------------------------------------------------------------------
__global__ __launch_bounds__(256)
void k_gemm_gatesx(const unsigned short* __restrict__ Xb,   // [MTOT x I] bf16
                   const unsigned short* __restrict__ Wt,   // [G x I] bf16
                   const float* __restrict__ bias,
                   float* __restrict__ C) {
  const int lane = threadIdx.x & 31;
  const int w    = threadIdx.x >> 5;
  const int wm   = w & 3;
  const int wn   = w >> 2;
  const int m0   = blockIdx.x * 128 + wm * 32;
  const int n0   = blockIdx.y * 128 + wn * 64;
  const int mrow = lane & 15;
  const int half = lane >> 4;

  v8f acc[2][4] = {};

  for (int k0 = 0; k0 < I_; k0 += 32) {
    v16bf a0 = load_a_bf(Xb, m0 + mrow,      I_, k0, lane);
    v16bf a1 = load_a_bf(Xb, m0 + 16 + mrow, I_, k0, lane);
#pragma unroll
    for (int j = 0; j < 4; ++j) {
      v16bf b = load_b_bf(Wt, n0 + j * 16 + mrow, I_, k0, lane);
      acc[0][j] = wmma_bf16(a0, b, acc[0][j]);
      acc[1][j] = wmma_bf16(a1, b, acc[1][j]);
    }
  }

#pragma unroll
  for (int i = 0; i < 2; ++i) {
#pragma unroll
    for (int j = 0; j < 4; ++j) {
      const int col = n0 + j * 16 + mrow;
      const float bv = bias[col];
#pragma unroll
      for (int g = 0; g < 8; ++g) {
        const int row = m0 + i * 16 + half * 8 + g;
        C[(size_t)row * G_ + col] = acc[i][j][g] + bv;
      }
    }
  }
}

// ---------------------------------------------------------------------------
// Phase 2: one GRU timestep.
//   g2 = h_prev @ hid_W + hid_b ;  r,z sigmoid ; hcand tanh ; blend.
// Block: 256 thr = 8 waves: wave = (m 0..3 over batch rows, nsub 0..1).
// Block covers M=64 (full batch) x 32 columns of H; 3 gate GEMMs fused.
// Grid: H/32 = 32 blocks per step. hid_W stream is L2-resident (6.3 MB bf16);
// explicit prefetch (global_prefetch_b8) hides L2 latency on the serial chain.
// ---------------------------------------------------------------------------
__global__ __launch_bounds__(256)
void k_gru_step(const unsigned short* __restrict__ h_bf,     // [B x H] bf16
                const unsigned short* __restrict__ hidWt,    // [G x H] bf16
                const float* __restrict__ gates_x,           // [B x S x G]
                const float* __restrict__ hid_b,             // [G]
                const float* __restrict__ h_prev,            // [B x H] f32
                float* __restrict__ h_next,                  // [B x H] f32
                unsigned short* __restrict__ h_next_bf,      // [B x H] bf16
                float* __restrict__ out_seq,                 // [B x S x H]
                int t) {
  const int lane = threadIdx.x & 31;
  const int w    = threadIdx.x >> 5;
  const int m    = w & 3;                    // batch sub-tile (16 rows)
  const int nsub = w >> 2;                   // 0..1
  const int c0   = blockIdx.x * 32 + nsub * 16;  // column base within H
  const int mrow = lane & 15;
  const int half = lane >> 4;
  const int row0 = m * 16;

  v8f ar = {}, az = {}, an = {};

  for (int k0 = 0; k0 < H_; k0 += 32) {
    // Prefetch next K-slab of the three weight rows this lane will read.
    if (k0 + 32 < H_) {
      __builtin_prefetch(hidWt + (size_t)(c0 + mrow) * H_ + k0 + 32, 0, 1);
      __builtin_prefetch(hidWt + (size_t)(H_ + c0 + mrow) * H_ + k0 + 32, 0, 1);
      __builtin_prefetch(hidWt + (size_t)(2 * H_ + c0 + mrow) * H_ + k0 + 32, 0, 1);
    }
    v16bf a  = load_a_bf(h_bf, row0 + mrow, H_, k0, lane);
    v16bf br = load_b_bf(hidWt,           c0 + mrow, H_, k0, lane);
    v16bf bz = load_b_bf(hidWt,   H_   +  c0 + mrow, H_, k0, lane);
    v16bf bn = load_b_bf(hidWt, 2*H_   +  c0 + mrow, H_, k0, lane);
    ar = wmma_bf16(a, br, ar);
    az = wmma_bf16(a, bz, az);
    an = wmma_bf16(a, bn, an);
  }

  const int col = c0 + mrow;
  const float hbr = hid_b[col];
  const float hbz = hid_b[H_ + col];
  const float hbn = hid_b[2 * H_ + col];

#pragma unroll
  for (int g = 0; g < 8; ++g) {
    const int b = row0 + half * 8 + g;  // batch index
    const float* gx = gates_x + (size_t)b * ((size_t)S_ * G_) + (size_t)t * G_;
    const float g2r = ar[g] + hbr;
    const float g2z = az[g] + hbz;
    const float g2n = an[g] + hbn;
    const float r  = 1.0f / (1.0f + __expf(-(gx[col]        + g2r)));
    const float z  = 1.0f / (1.0f + __expf(-(gx[H_ + col]   + g2z)));
    const float hc = tanhf(gx[2 * H_ + col] + r * g2n);
    const float hp = h_prev[(size_t)b * H_ + col];
    const float hn = (1.0f - z) * hc + z * hp;
    out_seq[(size_t)b * (S_ * H_) + (size_t)t * H_ + col] = hn;
    h_next[(size_t)b * H_ + col]    = hn;
    h_next_bf[(size_t)b * H_ + col] = f32_to_bf16_bits(hn);
  }
}

// ---------------------------------------------------------------------------
// Host-side orchestration
// ---------------------------------------------------------------------------
extern "C" void kernel_launch(void* const* d_in, const int* in_sizes, int n_in,
                              void* d_out, int out_size, void* d_ws, size_t ws_size,
                              hipStream_t stream) {
  const float* x     = (const float*)d_in[0];   // [B,S,I]
  const float* inp_W = (const float*)d_in[1];   // [I,3H]
  const float* hid_W = (const float*)d_in[2];   // [H,3H]
  const float* inp_b = (const float*)d_in[3];   // [3H]
  const float* hid_b = (const float*)d_in[4];   // [3H]
  float* out = (float*)d_out;                   // [B,S,H] ++ [B,H]

  char* ws = (char*)d_ws;
  size_t off = 0;
  float* gates_x = (float*)(ws + off);          off += (size_t)MTOT * G_ * 4;
  float* hA      = (float*)(ws + off);          off += (size_t)B_ * H_ * 4;
  float* hB      = (float*)(ws + off);          off += (size_t)B_ * H_ * 4;
  unsigned short* hAb = (unsigned short*)(ws + off); off += (size_t)B_ * H_ * 2;
  unsigned short* hBb = (unsigned short*)(ws + off); off += (size_t)B_ * H_ * 2;
  unsigned short* inpWt = (unsigned short*)(ws + off); off += (size_t)I_ * G_ * 2;
  unsigned short* hidWt = (unsigned short*)(ws + off); off += (size_t)H_ * G_ * 2;
  unsigned short* xB    = (unsigned short*)(ws + off); off += (size_t)MTOT * I_ * 2;
  (void)ws_size; (void)in_sizes; (void)n_in; (void)out_size;

  // Weight transpose + bf16 conversion (one pass each)
  {
    const int total = I_ * G_;
    const int blocks = (total + 255) / 256;
    k_transpose_bf16<<<blocks, 256, 0, stream>>>(inp_W, inpWt, I_, G_);
    k_transpose_bf16<<<blocks, 256, 0, stream>>>(hid_W, hidWt, H_, G_);
  }

  // x -> bf16, streaming pass (8 elems/thread)
  {
    const int n8 = MTOT * I_ / 8;
    k_convert_bf16_vec8<<<(n8 + 255) / 256, 256, 0, stream>>>(x, xB, n8);
  }

  // h0 = 0
  k_zero_state<<<(B_ * H_ + 255) / 256, 256, 0, stream>>>(hA, hAb, B_ * H_);

  // Phase 1: batched input projection (WMMA GEMM), 128x128 block tiles
  {
    dim3 grid(MTOT / 128, G_ / 128);
    k_gemm_gatesx<<<grid, 256, 0, stream>>>(xB, inpWt, inp_b, gates_x);
  }

  // Phase 2: sequential scan, one kernel per timestep (stream order = dep chain)
  float* hp = hA; float* hn = hB;
  unsigned short* hpb = hAb; unsigned short* hnb = hBb;
  for (int t = 0; t < S_; ++t) {
    k_gru_step<<<H_ / 32, 256, 0, stream>>>(hpb, hidWt, gates_x, hid_b,
                                            hp, hn, hnb, out, t);
    float* tf = hp; hp = hn; hn = tf;
    unsigned short* tb = hpb; hpb = hnb; hnb = tb;
  }

  // h_last lives in hp after the final swap
  k_copy_f32<<<(B_ * H_ + 255) / 256, 256, 0, stream>>>(
      hp, out + (size_t)B_ * S_ * H_, B_ * H_);
}
